// EmotionDetection_90434831384682
// MI455X (gfx1250) — matrix operand order repaired and checked by reading, not compiled, not run on hardware
//
#include <hip/hip_runtime.h>
#include <stdint.h>

// ---------------------------------------------------------------------------
// EmotionDetection RNN (3 fused GRUs + history attention) for MI455X/gfx1250.
// Persistent wave32 kernel, bf16 WMMA (v_wmma_f32_16x16x32_bf16), fp32 state.
// Round 4: explicit address_space(1) accesses now use a trivial ext-vector
// u32x4 (HIP's uint4 class ctor cannot bind an AS(1) reference on the host
// pass).  Goal: global_load_b128 on the GEMM hot path, not flat_load_b128.
// ---------------------------------------------------------------------------

#define U_   1024
#define B_   32
#define T_   256
#define LD3  3072
#define BU   (B_ * U_)
#define NWG  64

typedef unsigned short u16;
typedef __attribute__((ext_vector_type(16))) __bf16 v16bf;
typedef __attribute__((ext_vector_type(8)))  float  v8f;
typedef __attribute__((ext_vector_type(4)))  unsigned int u32x4;

union FragBF { v16bf v; u32x4 q[2]; };

#define GAS __attribute__((address_space(1)))

// Explicit global-address-space accessors: force global_load/global_store
// selection (flat would also consume LDS bandwidth and DScnt).
__device__ inline u32x4 gld4(const u16* p) { return *(const GAS u32x4*)(p); }
__device__ inline float gldf(const float* p) { return *(const GAS float*)(p); }
__device__ inline void  gstf(float* p, float v) { *(GAS float*)(p) = v; }
__device__ inline void  gsth(u16* p, u16 v) { *(GAS u16*)(p) = v; }

__device__ inline u16 f2bf(float f) {
  unsigned u = __float_as_uint(f);
  u += 0x7FFFu + ((u >> 16) & 1u);          // round-to-nearest-even
  return (u16)(u >> 16);
}

__device__ inline float sigm(float x) { return 1.f / (1.f + __expf(-x)); }

__device__ inline v8f v8zero() {
  v8f z;
#pragma unroll
  for (int i = 0; i < 8; ++i) z[i] = 0.f;
  return z;
}

__device__ inline v8f wmma_bf16(v16bf a, v16bf b, v8f c) {
  // (neg_a, A, neg_b, B, c_mod, C, reuse_a, reuse_b)
  return __builtin_amdgcn_wmma_f32_16x16x32_bf16(false, a, false, b, (short)0, c,
                                                 false, false);
}

// A-matrix 16x32 bf16 fragment per CDNA5 layout:
//   lanes 0-15 : row M = lane,    K = k0+{0..7, 16..23}
//   lanes 16-31: row M = lane-16, K = k0+{8..15, 24..31}
__device__ inline v16bf load_a(const u16* __restrict__ X, int m0, int k0,
                               int lane) {
  const u16* p = X + (size_t)(m0 + (lane & 15)) * U_ + k0 + ((lane & 16) ? 8 : 0);
  FragBF f;
  f.q[0] = gld4(p);
  f.q[1] = gld4(p + 16);
  return f.v;
}

// B-matrix 32x16 bf16 fragment: lane L holds column N = n0 + (L & 15),
// K = k0+{0..7, 16..23} (lanes 0-15) / k0+{8..15, 24..31} (lanes 16-31).
__device__ inline v16bf load_b(const u16* __restrict__ Wt, int ldn, int k0,
                               int n0, int lane) {
  const u16* p = Wt + (size_t)(k0 + ((lane & 16) ? 8 : 0)) * ldn + n0 + (lane & 15);
  union { v16bf v; u16 e[16]; } f;
#pragma unroll
  for (int s = 0; s < 16; ++s)
    f.e[s] = p[(size_t)((s < 8) ? s : s + 8) * ldn];
  return f.v;
}

// 6 gate accumulators (i_r,i_z,i_n,h_r,h_z,h_n) over kt in [kt0, kt1).
__device__ inline void gru_partial(const u16* __restrict__ Xb,
                                   const u16* __restrict__ Hb,
                                   const u16* __restrict__ Wih,
                                   const u16* __restrict__ Whh,
                                   int m0, int n0, int lane, int kt0, int kt1,
                                   v8f acc[6]) {
#pragma unroll
  for (int a = 0; a < 6; ++a) acc[a] = v8zero();
#pragma unroll 2
  for (int kt = kt0; kt < kt1; ++kt) {
    const int k0 = kt * 32;
    v16bf ax = load_a(Xb, m0, k0, lane);
    v16bf ah = load_a(Hb, m0, k0, lane);
    acc[0] = wmma_bf16(ax, load_b(Wih, LD3, k0, n0,          lane), acc[0]);
    acc[1] = wmma_bf16(ax, load_b(Wih, LD3, k0, n0 + U_,     lane), acc[1]);
    acc[2] = wmma_bf16(ax, load_b(Wih, LD3, k0, n0 + 2 * U_, lane), acc[2]);
    acc[3] = wmma_bf16(ah, load_b(Whh, LD3, k0, n0,          lane), acc[3]);
    acc[4] = wmma_bf16(ah, load_b(Whh, LD3, k0, n0 + U_,     lane), acc[4]);
    acc[5] = wmma_bf16(ah, load_b(Whh, LD3, k0, n0 + 2 * U_, lane), acc[5]);
  }
}

// Fused GRU phase, NS-way K split across NS partner waves of one workgroup.
// Partner waves stash partial accumulators in LDS; slice-0 wave reduces and
// runs the sigmoid/tanh gate math + stores.  One __syncthreads, no grid sync.
template <int NS>
__device__ void gru_phase(float* __restrict__ red, int jobslot, int slice,
                          const u16* __restrict__ Xb, const u16* __restrict__ Hb,
                          const u16* __restrict__ Wih, const u16* __restrict__ Whh,
                          const float* __restrict__ bih,
                          const float* __restrict__ bhh,
                          const float* __restrict__ Hf,
                          float* __restrict__ outF, u16* __restrict__ outB,
                          float* outF2, int ldF2, int m0, int n0, int lane) {
  v8f acc[6];
  const int KT = (U_ / 32) / NS;
  gru_partial(Xb, Hb, Wih, Whh, m0, n0, lane, slice * KT, slice * KT + KT, acc);
  float* rbase = red + (size_t)jobslot * (NS - 1) * 1536;
  if (slice != 0) {
    float* rp = rbase + (size_t)(slice - 1) * 1536;
#pragma unroll
    for (int a = 0; a < 6; ++a)
#pragma unroll
      for (int v = 0; v < 8; ++v) rp[a * 256 + v * 32 + lane] = acc[a][v];
  }
  __syncthreads();
  if (slice == 0) {
#pragma unroll
    for (int sl = 1; sl < NS; ++sl) {
      const float* rp = rbase + (size_t)(sl - 1) * 1536;
#pragma unroll
      for (int a = 0; a < 6; ++a)
#pragma unroll
        for (int v = 0; v < 8; ++v) acc[a][v] += rp[a * 256 + v * 32 + lane];
    }
    const int col = n0 + (lane & 15);
    const int rb  = m0 + ((lane & 16) ? 8 : 0);  // C/D layout: VGPR v -> M=rb+v
    const float bir = gldf(bih + col),          bhr = gldf(bhh + col);
    const float biz = gldf(bih + col + U_),     bhz = gldf(bhh + col + U_);
    const float bin = gldf(bih + col + 2 * U_), bhn = gldf(bhh + col + 2 * U_);
#pragma unroll
    for (int v = 0; v < 8; ++v) {
      const int r = rb + v;
      float rr = sigm((acc[0][v] + bir) + (acc[3][v] + bhr));
      float zz = sigm((acc[1][v] + biz) + (acc[4][v] + bhz));
      float nn = tanhf((acc[2][v] + bin) + rr * (acc[5][v] + bhn));
      float h  = gldf(Hf + (size_t)r * U_ + col);
      float o  = (1.f - zz) * nn + zz * h;
      gstf(outF + (size_t)r * U_ + col, o);
      gsth(outB + (size_t)r * U_ + col, f2bf(o));
      if (outF2) gstf(outF2 + (size_t)r * ldF2 + col, o);
    }
  }
}

// q = ut @ global_W, one 16x16 tile, 2-way K split.
__device__ void q_phase(float* __restrict__ red, int jobslot, int slice,
                        const u16* __restrict__ Xb, const u16* __restrict__ Wb,
                        float* __restrict__ qout, int m0, int n0, int lane) {
  v8f acc = v8zero();
#pragma unroll 2
  for (int kt = slice * 16; kt < slice * 16 + 16; ++kt)
    acc = wmma_bf16(load_a(Xb, m0, kt * 32, lane),
                    load_b(Wb, U_, kt * 32, n0, lane), acc);
  float* rp = red + (size_t)jobslot * 256;
  if (slice != 0) {
#pragma unroll
    for (int v = 0; v < 8; ++v) rp[v * 32 + lane] = acc[v];
  }
  __syncthreads();
  if (slice == 0) {
    const int col = n0 + (lane & 15);
    const int rb  = m0 + ((lane & 16) ? 8 : 0);
#pragma unroll
    for (int v = 0; v < 8; ++v)
      gstf(qout + (size_t)(rb + v) * U_ + col, acc[v] + rp[v * 32 + lane]);
  }
}

// Device-wide generation barrier (64 resident WGs; s_sleep spin).
__device__ inline void grid_sync(unsigned* bar, int tid) {
  __syncthreads();
  if (tid == 0) {
    __threadfence();
    unsigned g   = __atomic_load_n(&bar[1], __ATOMIC_ACQUIRE);
    unsigned old = __atomic_fetch_add(&bar[0], 1u, __ATOMIC_ACQ_REL);
    if (old == NWG - 1) {
      __atomic_store_n(&bar[0], 0u, __ATOMIC_RELAXED);
      __atomic_fetch_add(&bar[1], 1u, __ATOMIC_ACQ_REL);
    } else {
      while (__atomic_load_n(&bar[1], __ATOMIC_ACQUIRE) == g)
        __builtin_amdgcn_s_sleep(2);
    }
    __threadfence();
  }
  __syncthreads();
}

// ---------------------------------------------------------------------------

struct PrepParams {
  const float* wsrc[7];          // gW_ih,gW_hh,pW_ih,pW_hh,eW_ih,eW_hh,global_W
  u16* wdst[7];
  const float* inp;
  u16* xuT;                      // (T,B,U) bf16 of inputs, step-major
  u16* xg0;                      // xg buffer 0 = bf16(ut(0) + 0)
  float* zf0;  long long nzf0;   // zero fp32 (hist[0])
  float* zf1;  long long nzf1;   // zero fp32 (p0,p1,e0,e1)
  u16* zb;     long long nzb;    // zero bf16 state (g,e,p quad)
  unsigned* bar;
};

__global__ void prep_kernel(PrepParams P) {
  const long long NT  = 6LL * 3145728;   // 6 weight transposes (U x 3U)
  const long long NG  = 1048576;         // global_W convert (U x U)
  const long long NX  = 8388608;         // inputs -> bf16, step-major
  const long long NXG = BU;              // xg0
  const long long total = NT + NG + NX + NXG + P.nzf0 + P.nzf1 + P.nzb;
  const long long stride = (long long)gridDim.x * blockDim.x;
  for (long long idx = (long long)blockIdx.x * blockDim.x + threadIdx.x;
       idx < total; idx += stride) {
    long long r = idx;
    if (r < NT) {                                   // Wt[k][n] = W[n][k]
      int wsel = (int)(r / 3145728);
      int li   = (int)(r % 3145728);
      int k = li / LD3, n = li % LD3;
      gsth(P.wdst[wsel] + (size_t)k * LD3 + n,
           f2bf(gldf(P.wsrc[wsel] + (size_t)n * U_ + k)));
      continue;
    }
    r -= NT;
    if (r < NG) { gsth(P.wdst[6] + r, f2bf(gldf(P.wsrc[6] + r))); continue; }
    r -= NG;
    if (r < NX) {                                   // xuT[t][b][u] = inp[b][t][u]
      int t = (int)(r >> 15);
      int b = (int)((r >> 10) & 31);
      int u = (int)(r & 1023);
      gsth(P.xuT + r, f2bf(gldf(P.inp + ((size_t)b * T_ + t) * U_ + u)));
      continue;
    }
    r -= NX;
    if (r < NXG) {                                  // xg(0) = bf16(ut(0))
      int b = (int)(r >> 10), u = (int)(r & 1023);
      gsth(P.xg0 + r, f2bf(gldf(P.inp + (size_t)b * T_ * U_ + u)));
      continue;
    }
    r -= NXG;
    if (r < P.nzf0) { gstf(P.zf0 + r, 0.f); continue; }
    r -= P.nzf0;
    if (r < P.nzf1) { gstf(P.zf1 + r, 0.f); continue; }
    r -= P.nzf1;
    gsth(P.zb + r, 0);
  }
  if (blockIdx.x == 0 && threadIdx.x == 0) { P.bar[0] = 0u; P.bar[1] = 0u; }
}

struct RnnParams {
  const float* inp;
  const float *gbih, *gbhh, *pbih, *pbhh, *ebih, *ebhh;
  const u16 *gWih, *gWhh, *pWih, *pWhh, *eWih, *eWhh, *globW;
  const u16* xuT;
  float* hist;            // (T+1, B, U) fp32; [0] zeroed
  float* q;               // (B, U) fp32
  u16* xg;                // 2 x (B,U) bf16 : bf16(ut + p_cur), double buffered
  u16* xp;                // (B,U) bf16     : bf16(ut + ct)
  float* pf;              // 2 x (B,U) fp32 : p0, p1
  float* ef;              // 2 x (B,U) fp32 : e, double buffered
  u16* gbuf;              // 2 x (B,U) bf16 : g state
  u16* ebuf;              // 2 x (B,U) bf16 : e state
  u16* pb4;               // 4 x (B,U) bf16 : p state (2 slots x 2 generations)
  float* out;             // (B, T, H) fp32
  unsigned* bar;
};

__global__ __launch_bounds__(256) void rnn_persistent(RnnParams P) {
  const int tid  = threadIdx.x;
  const int lane = tid & 31;
  const int wave = tid >> 5;
  const int wg   = blockIdx.x;

  __shared__ float red[9216];        // 36 KB partial-accumulator scratch
  __shared__ float sc[T_ + 8];       // attention scores
  __shared__ float s_inv;

  for (int i = 0; i < T_; ++i) {
    const int s  = i & 1;
    const int s1 = s ^ 1;
    const int h2 = (i >> 1) & 1;

    const u16*   xg_r = P.xg   + (size_t)s  * BU;
    const u16*   gb_r = P.gbuf + (size_t)s  * BU;
    u16*         gb_w = P.gbuf + (size_t)s1 * BU;
    const u16*   pb_r = P.pb4  + (size_t)(s * 2 + h2) * BU;
    u16*         pb_w = P.pb4  + (size_t)(s * 2 + (h2 ^ 1)) * BU;
    float*       pf_c = P.pf   + (size_t)s  * BU;
    const float* ef_r = P.ef   + (size_t)s  * BU;
    float*       ef_w = P.ef   + (size_t)s1 * BU;
    const u16*   eb_r = P.ebuf + (size_t)s  * BU;
    u16*         eb_w = P.ebuf + (size_t)s1 * BU;

    // ---- Phase A: g_new GRU (WGs 0-31, K split 2) + q GEMM (WGs 32-63,
    //      K split 2).  512/512 waves busy. ----
    if (wg < 32) {
      const int pp = wave >> 1, slice = wave & 1;
      const int j  = wg * 4 + pp;                 // 0..127
      const int m0 = (j >> 6) << 4, n0 = (j & 63) << 4;
      gru_phase<2>(red, pp, slice, xg_r, gb_r, P.gWih, P.gWhh, P.gbih, P.gbhh,
                   P.hist + (size_t)i * BU,        // h  = g(i)  (== hist[i])
                   P.hist + (size_t)(i + 1) * BU,  // h' -> hist[i+1]
                   gb_w, nullptr, 0, m0, n0, lane);
    } else {
      const int pp = wave >> 1, slice = wave & 1;
      const int j  = (wg - 32) * 4 + pp;
      q_phase(red, pp, slice, P.xuT + (size_t)i * BU, P.globW, P.q,
              (j >> 6) << 4, (j & 63) << 4, lane);
    }
    grid_sync(P.bar, tid);

    // ---- Phase B: attention (WGs 0-31, batch b = wg); WGs 32-63 prebuild
    //      xg(i+1) = bf16(ut(i+1) + p[(i+1)&1])  (that slot is idle). ----
    if (wg < 32) {
      const int b = wg;
      const float* qrow = P.q + (size_t)b * U_;
      for (int t = wave; t <= i; t += 8) {
        const float* hrow = P.hist + ((size_t)t * B_ + b) * U_;
        float acc = 0.f;
        for (int u = lane; u < U_; u += 32) acc += gldf(hrow + u) * gldf(qrow + u);
        acc += __shfl_xor(acc, 16);
        acc += __shfl_xor(acc, 8);
        acc += __shfl_xor(acc, 4);
        acc += __shfl_xor(acc, 2);
        acc += __shfl_xor(acc, 1);
        if (lane == 0) sc[t] = acc;
      }
      __syncthreads();
      if (wave == 0) {
        float mx = -3.402823466e38f;
        for (int t = lane; t <= i; t += 32) mx = fmaxf(mx, sc[t]);
        for (int m = 16; m; m >>= 1) mx = fmaxf(mx, __shfl_xor(mx, m));
        float sum = 0.f;
        for (int t = lane; t <= i; t += 32) {
          float ex = __expf(sc[t] - mx);
          sc[t] = ex;
          sum += ex;
        }
        for (int m = 16; m; m >>= 1) sum += __shfl_xor(sum, m);
        if (lane == 0) s_inv = 1.f / sum;
      }
      __syncthreads();
      const float inv = s_inv;
      for (int u = tid; u < U_; u += 256) {
        float c = 0.f;
        for (int t = 0; t <= i; ++t)
          c += sc[t] * gldf(P.hist + ((size_t)t * B_ + b) * U_ + u);
        float xv = gldf(P.inp + ((size_t)b * T_ + i) * U_ + u) + c * inv;
        gsth(P.xp + (size_t)b * U_ + u, f2bf(xv));
      }
    } else if (i + 1 < T_) {
      const float* pnx = P.pf + (size_t)s1 * BU;   // written at step i-1
      u16* xgw = P.xg + (size_t)s1 * BU;
      for (int el = (wg - 32) * 256 + tid; el < BU; el += 32 * 256) {
        int b2 = el >> 10, u = el & 1023;
        gsth(xgw + el,
             f2bf(gldf(P.inp + ((size_t)b2 * T_ + (i + 1)) * U_ + u) +
                  gldf(pnx + el)));
      }
    }
    grid_sync(P.bar, tid);

    // ---- Phase C: p_new GRU, K split 4, all 64 WGs (512/512 waves). ----
    {
      const int jobslot = wave >> 2, slice = wave & 3;
      const int j  = wg * 2 + jobslot;            // 0..127
      const int m0 = (j >> 6) << 4, n0 = (j & 63) << 4;
      gru_phase<4>(red, jobslot, slice, P.xp, pb_r, P.pWih, P.pWhh,
                   P.pbih, P.pbhh, pf_c, pf_c, pb_w, nullptr, 0, m0, n0, lane);
    }
    grid_sync(P.bar, tid);

    // ---- Phase D: e_new GRU -> output row i, K split 4, all 64 WGs. ----
    {
      const int jobslot = wave >> 2, slice = wave & 3;
      const int j  = wg * 2 + jobslot;
      const int m0 = (j >> 6) << 4, n0 = (j & 63) << 4;
      gru_phase<4>(red, jobslot, slice, pb_w, eb_r, P.eWih, P.eWhh,
                   P.ebih, P.ebhh, ef_r, ef_w, eb_w,
                   P.out + (size_t)i * U_, T_ * U_, m0, n0, lane);
    }
    // No grid sync needed before next Phase A: A(i+1) reads xg[s1] (B(i),
    // 2 syncs ago), gbuf[s1]/hist[i+1] (A(i), 3 syncs ago) and writes
    // hist[i+2]/gbuf[s]/q -- all disjoint from Phase D(i) reads/writes.
  }
}

// ---------------------------------------------------------------------------

extern "C" void kernel_launch(void* const* d_in, const int* in_sizes, int n_in,
                              void* d_out, int out_size, void* d_ws,
                              size_t ws_size, hipStream_t stream) {
  (void)in_sizes; (void)n_in; (void)out_size; (void)ws_size;
  const float* inp  = (const float*)d_in[0];
  const float* gW   = (const float*)d_in[1];
  const float* gWih = (const float*)d_in[2];
  const float* gWhh = (const float*)d_in[3];
  const float* gbih = (const float*)d_in[4];
  const float* gbhh = (const float*)d_in[5];
  const float* pWih = (const float*)d_in[6];
  const float* pWhh = (const float*)d_in[7];
  const float* pbih = (const float*)d_in[8];
  const float* pbhh = (const float*)d_in[9];
  const float* eWih = (const float*)d_in[10];
  const float* eWhh = (const float*)d_in[11];
  const float* ebih = (const float*)d_in[12];
  const float* ebhh = (const float*)d_in[13];

  char* w = (char*)d_ws;
  auto bump = [&](size_t bytes) -> void* {
    void* p = (void*)w;
    w += (bytes + 255) & ~(size_t)255;
    return p;
  };

  u16* gWih_b  = (u16*)bump((size_t)U_ * LD3 * 2);
  u16* gWhh_b  = (u16*)bump((size_t)U_ * LD3 * 2);
  u16* pWih_b  = (u16*)bump((size_t)U_ * LD3 * 2);
  u16* pWhh_b  = (u16*)bump((size_t)U_ * LD3 * 2);
  u16* eWih_b  = (u16*)bump((size_t)U_ * LD3 * 2);
  u16* eWhh_b  = (u16*)bump((size_t)U_ * LD3 * 2);
  u16* globW_b = (u16*)bump((size_t)U_ * U_ * 2);
  u16* xuT     = (u16*)bump((size_t)T_ * BU * 2);
  float* hist  = (float*)bump((size_t)(T_ + 1) * BU * 4);
  float* q     = (float*)bump((size_t)BU * 4);
  u16* xg      = (u16*)bump((size_t)2 * BU * 2);
  u16* xp      = (u16*)bump((size_t)BU * 2);
  float* pf    = (float*)bump((size_t)2 * BU * 4);   // contiguous with ef
  float* ef    = (float*)bump((size_t)2 * BU * 4);
  u16* gbuf    = (u16*)bump((size_t)2 * BU * 2);     // contiguous with
  u16* ebuf    = (u16*)bump((size_t)2 * BU * 2);     // ebuf and pb4
  u16* pb4     = (u16*)bump((size_t)4 * BU * 2);
  unsigned* bar = (unsigned*)bump(256);

  PrepParams PP;
  PP.wsrc[0] = gWih; PP.wsrc[1] = gWhh;
  PP.wsrc[2] = pWih; PP.wsrc[3] = pWhh;
  PP.wsrc[4] = eWih; PP.wsrc[5] = eWhh;
  PP.wsrc[6] = gW;
  PP.wdst[0] = gWih_b; PP.wdst[1] = gWhh_b;
  PP.wdst[2] = pWih_b; PP.wdst[3] = pWhh_b;
  PP.wdst[4] = eWih_b; PP.wdst[5] = eWhh_b;
  PP.wdst[6] = globW_b;
  PP.inp = inp;
  PP.xuT = xuT;
  PP.xg0 = xg;                     // xg buffer 0 (used at step 0)
  PP.zf0 = hist;      PP.nzf0 = BU;            // hist[0] = 0
  PP.zf1 = pf;        PP.nzf1 = 4LL * BU;      // p0,p1,e0,e1 = 0
  PP.zb  = gbuf;      PP.nzb  = 8LL * BU;      // gbuf(2)+ebuf(2)+pb4(4) = 0
  PP.bar = bar;
  prep_kernel<<<4096, 256, 0, stream>>>(PP);

  RnnParams RP;
  RP.inp = inp;
  RP.gbih = gbih; RP.gbhh = gbhh;
  RP.pbih = pbih; RP.pbhh = pbhh;
  RP.ebih = ebih; RP.ebhh = ebhh;
  RP.gWih = gWih_b; RP.gWhh = gWhh_b;
  RP.pWih = pWih_b; RP.pWhh = pWhh_b;
  RP.eWih = eWih_b; RP.eWhh = eWhh_b;
  RP.globW = globW_b;
  RP.xuT = xuT;
  RP.hist = hist;
  RP.q = q;
  RP.xg = xg;
  RP.xp = xp;
  RP.pf = pf;
  RP.ef = ef;
  RP.gbuf = gbuf;
  RP.ebuf = ebuf;
  RP.pb4 = pb4;
  RP.out = (float*)d_out;
  RP.bar = bar;
  rnn_persistent<<<NWG, 256, 0, stream>>>(RP);
}
